// ToMeBlock_26001732010505
// MI455X (gfx1250) — matrix operand order repaired and checked by reading, not compiled, not run on hardware
//
#include <hip/hip_runtime.h>

// ---- CDNA5 WMMA types ----
typedef __attribute__((ext_vector_type(16))) _Float16 v16h;
typedef __attribute__((ext_vector_type(8)))  _Float16 v8h;
typedef __attribute__((ext_vector_type(4)))  _Float16 v4h;
typedef __attribute__((ext_vector_type(8)))  float    v8f;

// Problem constants (from reference setup_inputs)
#define BATCH 64
#define TT    577
#define CC    1024
#define NA    289          // even tokens (rows of 'a')
#define NApad 304          // padded to multiple of 16 (19 tiles)
#define NB    288          // odd tokens  (cols of scores)
#define MT    19           // row tiles per batch
#define NTI   18           // col tiles per batch

union Frag16 { v16h v; v8h h[2]; };

// -----------------------------------------------------------------------------
// Kernel 1: row-normalize x into f16 A (even tokens, zero-padded to 304 rows)
// and f16 B (odd tokens). One wave (32 lanes) per row.
// -----------------------------------------------------------------------------
__global__ __launch_bounds__(32)
void metric_f16_kernel(const float* __restrict__ x,
                       _Float16* __restrict__ A,
                       _Float16* __restrict__ Bm)
{
    const int slot = blockIdx.x % (NApad + NB);   // 0..591
    const int b    = blockIdx.x / (NApad + NB);
    const int lane = threadIdx.x;

    const bool isA = slot < NApad;
    const int  row = isA ? slot : slot - NApad;
    const int  tok = isA ? 2 * row : 2 * row + 1;

    _Float16* dst = isA ? (A  + ((size_t)b * NApad + row) * CC)
                        : (Bm + ((size_t)b * NB    + row) * CC);

    if (isA && tok >= TT) {              // pad rows 289..303 -> zeros
        v8h z;
        #pragma unroll
        for (int t = 0; t < 8; ++t) z[t] = (_Float16)0.0f;
        for (int i = lane * 8; i < CC; i += 32 * 8)
            *(v8h*)(dst + i) = z;
        return;
    }

    const float* src = x + ((size_t)b * TT + tok) * CC;

    float ss = 0.0f;
    for (int i = lane * 4; i < CC; i += 32 * 4) {
        float4 v = *(const float4*)(src + i);
        ss += v.x * v.x + v.y * v.y + v.z * v.z + v.w * v.w;
    }
    #pragma unroll
    for (int m = 16; m >= 1; m >>= 1)
        ss += __shfl_xor(ss, m, 32);

    const float inv = rsqrtf(ss);

    for (int i = lane * 4; i < CC; i += 32 * 4) {
        float4 v = *(const float4*)(src + i);
        v4h h;
        h[0] = (_Float16)(v.x * inv);
        h[1] = (_Float16)(v.y * inv);
        h[2] = (_Float16)(v.z * inv);
        h[3] = (_Float16)(v.w * inv);
        *(v4h*)(dst + i) = h;
    }
}

// -----------------------------------------------------------------------------
// Kernel 2: per batch, scores = A (289x1024) * B^T (1024x288) via
// v_wmma_f32_16x16x32_f16, with the row-argmax fused into the epilogue.
// One wave computes a 16-row strip; loops over 18 column tiles; K-loop 32 WMMAs.
// Writes node_idx only (node_max / argsort are provably unnecessary: r = 288 of
// 289, and the protected -inf row is always the single unmerged token).
// -----------------------------------------------------------------------------
__global__ __launch_bounds__(32)
void score_argmax_kernel(const _Float16* __restrict__ A,
                         const _Float16* __restrict__ Bm,
                         int* __restrict__ node_idx)
{
    const int wid  = blockIdx.x;          // 0 .. BATCH*MT-1
    const int b    = wid / MT;
    const int mt   = wid % MT;            // row tile
    const int lane = threadIdx.x;
    const int mloc = lane & 15;
    const int hi   = lane >> 4;

    // A fragment rows: lanes 0-15 and 16-31 both cover M=0..15 (different K).
    const _Float16* Arow = A + ((size_t)b * NApad + mt * 16 + mloc) * CC;

    float bestv[8];
    int   bestn[8];
    #pragma unroll
    for (int r = 0; r < 8; ++r) { bestv[r] = -3.4e38f; bestn[r] = 0; }

    for (int nt = 0; nt < NTI; ++nt) {
        // B operand column N = mloc  ->  row (nt*16+mloc) of Bm (B = Bm^T).
        const _Float16* Brow = Bm + ((size_t)b * NB + nt * 16 + mloc) * CC;

        v8f c = {};
        #pragma unroll 2
        for (int k0 = 0; k0 < CC; k0 += 32) {
            Frag16 fa, fb;
            // 16-bit A 16x32 layout: lane(hi) selects K half-groups of 8.
            fa.h[0] = *(const v8h*)(Arow + k0 +      hi * 8);
            fa.h[1] = *(const v8h*)(Arow + k0 + 16 + hi * 8);
            // 16-bit B 32x16 layout: lane(hi) selects K half-groups of 8 (same as A).
            fb.h[0] = *(const v8h*)(Brow + k0 +      hi * 8);
            fb.h[1] = *(const v8h*)(Brow + k0 + 16 + hi * 8);
            c = __builtin_amdgcn_wmma_f32_16x16x32_f16(
                    false, fa.v, false, fb.v, (short)0, c, false, false);
        }

        const int ncol = nt * 16 + mloc;
        #pragma unroll
        for (int r = 0; r < 8; ++r) {
            float v = c[r];
            if (v > bestv[r]) { bestv[r] = v; bestn[r] = ncol; }
        }
    }

    // C layout: lanes 0-15 VGPR r -> M=r ; lanes 16-31 VGPR r -> M=8+r.
    // Reduce (max, argmin-index-on-tie) across the 16 lanes of each half.
    #pragma unroll
    for (int r = 0; r < 8; ++r) {
        float v = bestv[r];
        int   n = bestn[r];
        #pragma unroll
        for (int m = 8; m >= 1; m >>= 1) {
            float ov = __shfl_xor(v, m, 16);
            int   on = __shfl_xor(n, m, 16);
            if (ov > v || (ov == v && on < n)) { v = ov; n = on; }
        }
        if (mloc == 0) {
            const int mrow = mt * 16 + hi * 8 + r;
            node_idx[b * NApad + mrow] = n;
        }
    }
}

// -----------------------------------------------------------------------------
// Kernel 3: deterministic, atomic-free merge + divide.
// One 256-thread block per output row (B x 289 rows). Row 0 = even token 0.
// Row 1+j = (x_odd[j] + sum_{i: idx[i]==j} x_even[i]) / (1 + count).
// Fixed ascending-i accumulation order => bitwise-identical across replays.
// -----------------------------------------------------------------------------
__global__ __launch_bounds__(256)
void merge_kernel(const float* __restrict__ x,
                  const int* __restrict__ node_idx,
                  float* __restrict__ out)
{
    const int orow = blockIdx.x % NA;
    const int b    = blockIdx.x / NA;
    const int tid  = threadIdx.x;

    const float* xb = x   + (size_t)b * TT * CC;
    float*       o  = out + ((size_t)b * NA + orow) * CC;
    const int    c  = tid * 4;           // 256 threads * float4 = 1024

    if (orow == 0) {                     // protected token (even token 0)
        *(float4*)(o + c) = *(const float4*)(xb + c);
        return;
    }

    __shared__ int sidx[NA];
    for (int i = tid; i < NA; i += 256)
        sidx[i] = node_idx[b * NApad + i];
    __syncthreads();

    const int j = orow - 1;
    float4 acc = *(const float4*)(xb + (size_t)(2 * j + 1) * CC + c);
    int count = 1;

    for (int i = 1; i < NA; ++i) {       // uniform branch across the block
        if (sidx[i] == j) {
            float4 v = *(const float4*)(xb + (size_t)(2 * i) * CC + c);
            acc.x += v.x; acc.y += v.y; acc.z += v.z; acc.w += v.w;
            ++count;
        }
    }

    const float inv = 1.0f / (float)count;
    acc.x *= inv; acc.y *= inv; acc.z *= inv; acc.w *= inv;
    *(float4*)(o + c) = acc;
}

// -----------------------------------------------------------------------------
// Host launcher
// -----------------------------------------------------------------------------
static inline size_t align256(size_t v) { return (v + 255) & ~(size_t)255; }

extern "C" void kernel_launch(void* const* d_in, const int* in_sizes, int n_in,
                              void* d_out, int out_size, void* d_ws, size_t ws_size,
                              hipStream_t stream)
{
    (void)in_sizes; (void)n_in; (void)out_size; (void)ws_size;

    const float* x   = (const float*)d_in[0];
    float*       out = (float*)d_out;

    char* ws = (char*)d_ws;
    const size_t szA   = (size_t)BATCH * NApad * CC * sizeof(_Float16); // ~39.8 MB
    const size_t szB   = (size_t)BATCH * NB    * CC * sizeof(_Float16); // ~37.7 MB
    _Float16* A   = (_Float16*)(ws);
    _Float16* Bm  = (_Float16*)(ws + align256(szA));
    int*      idx = (int*)     (ws + align256(szA) + align256(szB));

    // 1) normalize rows -> f16 staging buffers (WMMA operand layout friendly)
    metric_f16_kernel<<<BATCH * (NApad + NB), 32, 0, stream>>>(x, A, Bm);

    // 2) fused WMMA GEMM + row argmax
    score_argmax_kernel<<<BATCH * MT, 32, 0, stream>>>(A, Bm, idx);

    // 3) deterministic merge + divide
    merge_kernel<<<BATCH * NA, 256, 0, stream>>>(x, idx, out);
}
